// PyTorchFMoE_FC_40132174414265
// MI455X (gfx1250) — matrix operand order repaired and checked by reading, not compiled, not run on hardware
//
#include <hip/hip_runtime.h>

// MoE FC (2 experts, top-1) for MI455X / gfx1250.
// v_wmma_f32_16x16x32_f16 GEMM fed by double-buffered LDS tiles filled with
// CDNA5 async copies (global_load_async_to_lds_b128, ASYNCcnt-tracked).
// An inline-asm register pin forces all fragment ds_loads to complete before
// the WMMA sequence: one LDS round-trip per K-slice, WMMAs back-to-back.

typedef __attribute__((ext_vector_type(16))) _Float16 v16h;
typedef __attribute__((ext_vector_type(8)))  _Float16 v8h;
typedef __attribute__((ext_vector_type(8)))  float    v8f;

// 16-byte global->LDS async copy, per-lane addresses (ISA 10: async LDS ops).
__device__ __forceinline__ void async_cp16(void* ldst, const void* gsrc) {
    uint32_t loff = (uint32_t)(uintptr_t)(__attribute__((address_space(3))) void*)ldst;
    unsigned long long ga = (unsigned long long)(uintptr_t)gsrc;
    asm volatile("global_load_async_to_lds_b128 %0, %1, off"
                 :: "v"(loff), "v"(ga) : "memory");
}
__device__ __forceinline__ void wait_async0() {
    asm volatile("s_wait_asynccnt 0x0" ::: "memory");
}

// ---------------------------------------------------------------- init
__global__ void init_cnt(unsigned* cnt) {
    if (threadIdx.x < 2) cnt[threadIdx.x] = 0u;
}

// ---------------------------------------------------------------- weights
// wq0[i] = f16(w0[i]); wq1[i] = f16(sign(w)*2^clamp(rint(log2|w|+eps),-15,0))
// (power-of-two weights are EXACT in f16; WMMA preserves denorms)
__global__ void prep_weights(const float* __restrict__ w0,
                             const float* __restrict__ w1,
                             _Float16* __restrict__ wq0,
                             _Float16* __restrict__ wq1, int n) {
    int i = blockIdx.x * blockDim.x + threadIdx.x;
    if (i >= n) return;
    wq0[i] = (_Float16)w0[i];
    float w  = w1[i];
    float sh = rintf(log2f(fabsf(w) + 1e-12f));   // rintf == round-half-even
    sh = fminf(fmaxf(sh, -15.0f), 0.0f);
    float q = exp2f(sh);
    float s = (w > 0.0f) ? q : ((w < 0.0f) ? -q : 0.0f);
    wq1[i] = (_Float16)s;
}

// ---------------------------------------------------------------- gate + gather
// One wave32 per token: logits, top-1, atomic slot, convert row to f16.
// Expert 0 rows pack from the front of xg, expert 1 rows from the back.
__global__ __launch_bounds__(256) void gate_gather(
        const float* __restrict__ x, const float* __restrict__ wg,
        _Float16* __restrict__ xg, int* __restrict__ rowmap,
        unsigned* __restrict__ cnt, int T, int C) {
    int tok  = (int)((blockIdx.x * blockDim.x + threadIdx.x) >> 5);
    int lane = threadIdx.x & 31;
    if (tok >= T) return;

    const float* xr = x + (size_t)tok * C;
    float s0 = 0.0f, s1 = 0.0f;
    for (int c = lane; c < C; c += 32) {
        float v = xr[c];
        s0 = fmaf(v, wg[c], s0);
        s1 = fmaf(v, wg[C + c], s1);
    }
    for (int off = 16; off > 0; off >>= 1) {
        s0 += __shfl_xor(s0, off, 32);
        s1 += __shfl_xor(s1, off, 32);
    }
    int e = (s1 > s0) ? 1 : 0;                 // jnp.argmax: ties -> expert 0

    int pos = 0;
    if (lane == 0) pos = (int)atomicAdd(&cnt[e], 1u);
    pos = __shfl(pos, 0, 32);
    int p = e ? (T - 1 - pos) : pos;           // two-ended packing
    if (lane == 0) rowmap[p] = tok;

    _Float16* dst = xg + (size_t)p * C;
    if (e == 0) {
        for (int c = lane; c < C; c += 32) dst[c] = (_Float16)xr[c];
    } else {
        // DeepShift fixed-point: clip(rint(x*2^16))/2^16
        for (int c = lane; c < C; c += 32) {
            float v = xr[c];
            float r = rintf(v * 65536.0f);
            r = fminf(fmaxf(r, -2147483648.0f), 2147483648.0f);
            dst[c] = (_Float16)(r * (1.0f / 65536.0f));
        }
    }
}

// ---------------------------------------------------------------- WMMA GEMM
// grid = (H/128, ceil(T/64), 2 experts), 256 threads = 8 waves.
// Block tile 64(M) x 128(N); wave (mi,ni) owns 16x64 -> 4 v8f accumulators.
// K-slices (32 wide) double-buffered in LDS via async copies.
//
// LDS layouts (80B padded row stride -> conflict-free ds_load_b128):
//   sA[buf][row(64)][40 halves]  : row r, chunk j (j=0..3) at r*40 + j*8 = K j*8..j*8+7
//   sB[buf][col(128)][40 halves] : col c, khalf kh, part pt at c*40 + kh*16 + pt*8
__global__ __launch_bounds__(256) void moe_wmma_gemm(
        const _Float16* __restrict__ Wq,    // [2][H][C] f16
        const _Float16* __restrict__ Xg,    // [T][C] gathered f16
        const int* __restrict__ rowmap,     // [T] gathered row -> token
        const unsigned* __restrict__ cnt,   // [2]
        const float* __restrict__ b0, const float* __restrict__ b1,
        float* __restrict__ out, int T, int C, int H) {
    const int e    = blockIdx.z;
    const int cntE = (int)cnt[e];
    const int rowBase = blockIdx.y * 64;
    if (rowBase >= cntE) return;               // uniform early exit
    const int colBase = blockIdx.x * 128;

    __shared__ __align__(16) _Float16 sA[2][64 * 40];
    __shared__ __align__(16) _Float16 sB[2][128 * 40];

    const int wave = threadIdx.x >> 5;
    const int lane = threadIdx.x & 31;
    const int mi = wave & 3, ni = wave >> 2;
    const int l16 = lane & 15, hsel = lane >> 4;

    const _Float16* Wbase = Wq + (size_t)e * H * C;
    const int ITERS = C / 32;

    // ---- async fill of one K-slice (all 256 threads; 16B chunks)
    auto fill = [&](int buf, int kk) {
        {   // A tile: 64 rows x 32 halves = 256 x 16B, one chunk/thread
            int row = threadIdx.x >> 2, j = threadIdx.x & 3;
            int lr = rowBase + row; if (lr > T - 1) lr = T - 1;
            int p  = e ? (T - 1 - lr) : lr;
            const _Float16* src = Xg + (size_t)p * C + kk + j * 8;
            async_cp16(&sA[buf][row * 40 + j * 8], src);
        }
#pragma unroll
        for (int i = 0; i < 2; ++i) {  // B tile: 128 cols x 32 halves = 512 x 16B
            int cid  = (threadIdx.x << 1) | i;
            int colw = cid >> 2, kh = (cid >> 1) & 1, pt = cid & 1;
            const _Float16* src = Wbase + (size_t)(colBase + colw) * C
                                        + kk + kh * 16 + pt * 8;
            async_cp16(&sB[buf][colw * 40 + kh * 16 + pt * 8], src);
        }
    };

    v8f acc[4] = {v8f{0}, v8f{0}, v8f{0}, v8f{0}};

    fill(0, 0);
    for (int it = 0; it < ITERS; ++it) {
        const int cur = it & 1;
        wait_async0();        // own fills (in-order) complete
        __syncthreads();      // everyone's fills visible; prev readers done
        if (it + 1 < ITERS) fill(1 - cur, (it + 1) * 32);

        // ---- issue ALL fragment loads.
        // A fragment (contiguous K order, same as B fragments):
        // lane row = l16; K = {hsel*16..+15}
        const _Float16* aP = &sA[cur][(mi * 16 + l16) * 40];
        v8h alo = *(const v8h*)(aP + hsel * 16);
        v8h ahi = *(const v8h*)(aP + hsel * 16 + 8);

        v8h blo0, bhi0, blo1, bhi1, blo2, bhi2, blo3, bhi3;
        {
            const _Float16* bP0 = &sB[cur][(ni * 64 +  0 + l16) * 40 + hsel * 16];
            const _Float16* bP1 = &sB[cur][(ni * 64 + 16 + l16) * 40 + hsel * 16];
            const _Float16* bP2 = &sB[cur][(ni * 64 + 32 + l16) * 40 + hsel * 16];
            const _Float16* bP3 = &sB[cur][(ni * 64 + 48 + l16) * 40 + hsel * 16];
            blo0 = *(const v8h*)(bP0); bhi0 = *(const v8h*)(bP0 + 8);
            blo1 = *(const v8h*)(bP1); bhi1 = *(const v8h*)(bP1 + 8);
            blo2 = *(const v8h*)(bP2); bhi2 = *(const v8h*)(bP2 + 8);
            blo3 = *(const v8h*)(bP3); bhi3 = *(const v8h*)(bP3 + 8);
        }

        // Pin all fragments live here: forces every ds_load to complete at this
        // point (single s_wait_dscnt), then the 4 WMMAs issue back-to-back.
        asm volatile("" : "+v"(alo), "+v"(ahi),
                          "+v"(blo0), "+v"(bhi0), "+v"(blo1), "+v"(bhi1),
                          "+v"(blo2), "+v"(bhi2), "+v"(blo3), "+v"(bhi3));

        v16h a = __builtin_shufflevector(alo, ahi,
                 0,1,2,3,4,5,6,7,8,9,10,11,12,13,14,15);
        v16h b0v = __builtin_shufflevector(blo0, bhi0,
                 0,1,2,3,4,5,6,7,8,9,10,11,12,13,14,15);
        v16h b1v = __builtin_shufflevector(blo1, bhi1,
                 0,1,2,3,4,5,6,7,8,9,10,11,12,13,14,15);
        v16h b2v = __builtin_shufflevector(blo2, bhi2,
                 0,1,2,3,4,5,6,7,8,9,10,11,12,13,14,15);
        v16h b3v = __builtin_shufflevector(blo3, bhi3,
                 0,1,2,3,4,5,6,7,8,9,10,11,12,13,14,15);

        acc[0] = __builtin_amdgcn_wmma_f32_16x16x32_f16(
                     false, a, false, b0v, (short)0, acc[0], false, false);
        acc[1] = __builtin_amdgcn_wmma_f32_16x16x32_f16(
                     false, a, false, b1v, (short)0, acc[1], false, false);
        acc[2] = __builtin_amdgcn_wmma_f32_16x16x32_f16(
                     false, a, false, b2v, (short)0, acc[2], false, false);
        acc[3] = __builtin_amdgcn_wmma_f32_16x16x32_f16(
                     false, a, false, b3v, (short)0, acc[3], false, false);
    }

    // D layout: VGPR j, lane -> (M = j + 8*hsel, N = l16). Bias + scatter.
    const float* bias = e ? b1 : b0;
    const int col0 = colBase + ni * 64 + l16;
    int tok[8];
#pragma unroll
    for (int j = 0; j < 8; ++j) {
        int lr = rowBase + mi * 16 + j + 8 * hsel;
        tok[j] = (lr < cntE) ? rowmap[e ? (T - 1 - lr) : lr] : -1;
    }
#pragma unroll
    for (int f = 0; f < 4; ++f) {
        int col = col0 + f * 16;
        float bv = bias[col];
#pragma unroll
        for (int j = 0; j < 8; ++j) {
            if (tok[j] >= 0)
                out[(size_t)tok[j] * H + col] = acc[f][j] + bv;
        }
    }
}

// ---------------------------------------------------------------- launch
extern "C" void kernel_launch(void* const* d_in, const int* in_sizes, int n_in,
                              void* d_out, int out_size, void* d_ws, size_t ws_size,
                              hipStream_t stream) {
    const float* inp    = (const float*)d_in[0];
    const float* w_gate = (const float*)d_in[1];
    const float* w0     = (const float*)d_in[2];
    const float* b0     = (const float*)d_in[3];
    const float* w1     = (const float*)d_in[4];
    const float* b1     = (const float*)d_in[5];
    float* out = (float*)d_out;

    const int C = in_sizes[1] / 2;     // w_gate is [2, C]
    const int H = in_sizes[3];         // b0 is [H]
    const int T = in_sizes[0] / C;     // inp is [B*N, C] flat

    // workspace: wq[2*H*C] f16 | xg[T*C] f16 | rowmap[T] i32 | cnt[2] u32
    char* ws = (char*)d_ws;
    size_t off = 0;
    _Float16* wq = (_Float16*)(ws + off); off += (size_t)2 * H * C * sizeof(_Float16);
    _Float16* xg = (_Float16*)(ws + off); off += (size_t)T * C * sizeof(_Float16);
    int* rowmap  = (int*)(ws + off);      off += (size_t)T * sizeof(int);
    off = (off + 255) & ~(size_t)255;
    unsigned* cnt = (unsigned*)(ws + off);

    init_cnt<<<1, 32, 0, stream>>>(cnt);

    const int nW = H * C;
    prep_weights<<<(nW + 255) / 256, 256, 0, stream>>>(w0, w1, wq, wq + (size_t)H * C, nW);

    gate_gather<<<(T + 7) / 8, 256, 0, stream>>>(inp, w_gate, xg, rowmap, cnt, T, C);

    dim3 grid((H + 127) / 128, (T + 63) / 64, 2);
    moe_wmma_gemm<<<grid, 256, 0, stream>>>(wq, xg, rowmap, cnt, b0, b1, out, T, C, H);
}
